// GATNet_7713761263891
// MI455X (gfx1250) — matrix operand order repaired and checked by reading, not compiled, not run on hardware
//
#include <hip/hip_runtime.h>

#define NEG_SLOPE 0.2f
#define EPSF 1e-16f

typedef __attribute__((ext_vector_type(16))) __bf16 v16bf;
typedef __attribute__((ext_vector_type(8)))  float  v8f;

union FragAB { v16bf v; uint4 q[2]; };

static __device__ __forceinline__ unsigned short f2bf(float f) {
  unsigned u = __float_as_uint(f);
  u += 0x7FFFu + ((u >> 16) & 1u);          // round-to-nearest-even
  return (unsigned short)(u >> 16);
}
// monotonic float<->uint encoding so atomicMax(uint) == float max
static __device__ __forceinline__ unsigned encF(float f) {
  unsigned b = __float_as_uint(f);
  return (b & 0x80000000u) ? ~b : (b | 0x80000000u);
}
static __device__ __forceinline__ float decF(unsigned u) {
  unsigned b = (u & 0x80000000u) ? (u ^ 0x80000000u) : ~u;
  return __uint_as_float(b);
}
static __device__ __forceinline__ float eluf(float x) {
  return x > 0.f ? x : (__expf(x) - 1.f);
}

// ---------------- conversion kernels ----------------
__global__ void k_f32_to_bf16(const float* __restrict__ in,
                              unsigned short* __restrict__ out, int n4) {
  int i = blockIdx.x * blockDim.x + threadIdx.x;
  if (i >= n4) return;
  float4 v = ((const float4*)in)[i];
  ushort4 o; o.x = f2bf(v.x); o.y = f2bf(v.y); o.z = f2bf(v.z); o.w = f2bf(v.w);
  ((ushort4*)out)[i] = o;
}

// W[K][Ncol] row-major  ->  WT[Ncol][K] bf16 (column-major B for WMMA frags)
__global__ void k_transpose_bf16(const float* __restrict__ W,
                                 unsigned short* __restrict__ WT, int K, int Ncol) {
  int i = blockIdx.x * blockDim.x + threadIdx.x;
  if (i >= K * Ncol) return;
  int k = i / Ncol, n = i - k * Ncol;
  WT[n * K + k] = f2bf(W[i]);
}

// ---------------- WMMA GEMM: C[M,Ncol] = A[M,256] * B[256,Ncol] ----------------
// A: bf16 row-major [M x 256]; BT: bf16 [Ncol x 256] (B column-major).
// One wave per 16x16 output tile, K-loop of 16x16x32 bf16 WMMA, f32 accumulate.
__global__ void k_wmma_gemm_bf16(const unsigned short* __restrict__ A,
                                 const unsigned short* __restrict__ BT,
                                 float* __restrict__ C,
                                 int m_tiles, int n_tiles) {
  const int K = 256;
  int tile = blockIdx.x * blockDim.y + threadIdx.y;
  if (tile >= m_tiles * n_tiles) return;          // wave-uniform: EXEC stays all-1
  int tm = tile / n_tiles, tn = tile - tm * n_tiles;
  int lane = threadIdx.x;                          // 0..31
  int idx16 = lane & 15, half = lane >> 4;

  const unsigned short* arow = A  + (size_t)(tm * 16 + idx16) * K; // A row M=idx16
  const unsigned short* bcol = BT + (size_t)(tn * 16 + idx16) * K; // B col N=idx16
  v8f acc = {};
  for (int kb = 0; kb < K; kb += 32) {
    FragAB a, b;
    // A 16x32 bf16 layout: elems 0-7 -> K = kb + half*8 + j ; elems 8-15 -> +16
    a.q[0] = *(const uint4*)(arow + kb + half * 8);
    a.q[1] = *(const uint4*)(arow + kb + 16 + half * 8);
    // B 32x16 bf16 layout: lane holds col, elems 0-7 -> K = kb + half*8 + j ; elems 8-15 -> +16
    b.q[0] = *(const uint4*)(bcol + kb + half * 8);
    b.q[1] = *(const uint4*)(bcol + kb + 16 + half * 8);
    acc = __builtin_amdgcn_wmma_f32_16x16x32_bf16(
        false, a.v, false, b.v, (short)0, acc, false, false);
  }
  // D layout: VGPR r, lane -> row = r + 8*half, col = idx16
  int ncols = n_tiles * 16;
  for (int r = 0; r < 8; ++r)
    C[(size_t)(tm * 16 + r + 8 * half) * ncols + tn * 16 + idx16] = acc[r];
}

// ---------------- attention score dots: a[n,h] = sum_c h[n,h,c]*att[h,c] ----------------
__global__ void k_attn_scores(const float* __restrict__ h,
                              const float* __restrict__ att_s,
                              const float* __restrict__ att_d,
                              float* __restrict__ as_, float* __restrict__ ad_,
                              int N, int H, int C) {
  int i = blockIdx.x * blockDim.x + threadIdx.x;
  if (i >= N * H) return;
  int n = i / H, hh = i - n * H;
  const float* hp = h + (size_t)n * H * C + hh * C;
  const float* sp = att_s + hh * C;
  const float* dp = att_d + hh * C;
  float s = 0.f, d = 0.f;
  for (int c = 0; c < C; ++c) { float v = hp[c]; s += v * sp[c]; d += v * dp[c]; }
  as_[i] = s; ad_[i] = d;
}

// ---------------- edge pass A: segment max (encoded uint atomicMax) ----------------
__global__ void k_edge_max(const int* __restrict__ ei, int E, int N,
                           const float* __restrict__ as_, const float* __restrict__ ad_,
                           unsigned* __restrict__ m, int H) {
  int idx = blockIdx.x * blockDim.x + threadIdx.x;
  int tot = (E + N) * H;
  if (idx >= tot) return;
  int e = idx / H, hh = idx - e * H;
  int s, dn;
  if (e < E) { s = ei[e]; dn = ei[E + e]; } else { s = dn = e - E; }
  float sc = as_[s * H + hh] + ad_[dn * H + hh];
  sc = sc > 0.f ? sc : NEG_SLOPE * sc;
  atomicMax(&m[dn * H + hh], encF(sc));
}

// ---------------- edge pass B: ex = exp(score - max); segment sum; store ex ----------------
__global__ void k_edge_expsum(const int* __restrict__ ei, int E, int N,
                              const float* __restrict__ as_, const float* __restrict__ ad_,
                              const unsigned* __restrict__ m, float* __restrict__ ssum,
                              float* __restrict__ exv, int H) {
  int idx = blockIdx.x * blockDim.x + threadIdx.x;
  int tot = (E + N) * H;
  if (idx >= tot) return;
  int e = idx / H, hh = idx - e * H;
  int s, dn;
  if (e < E) { s = ei[e]; dn = ei[E + e]; } else { s = dn = e - E; }
  float sc = as_[s * H + hh] + ad_[dn * H + hh];
  sc = sc > 0.f ? sc : NEG_SLOPE * sc;
  float ex = __expf(sc - decF(m[dn * H + hh]));
  atomicAdd(&ssum[dn * H + hh], ex);
  exv[idx] = ex;
}

// ---------------- edge scatter: agg[dst] += h[src] * alpha (float4 + 4 f32 atomics) ----------------
__global__ void k_edge_scatter(const int* __restrict__ ei, int E, int N,
                               const float* __restrict__ exv, const float* __restrict__ ssum,
                               const float* __restrict__ h, float* __restrict__ agg,
                               int H, int C) {
  int F = H * C;
  int chunks = F >> 2;
  int idx = blockIdx.x * blockDim.x + threadIdx.x;
  int e = idx / chunks;
  if (e >= E + N) return;
  int t = idx - e * chunks;
  int c0 = t * 4, hh = c0 / C;
  int s, dn;
  if (e < E) { s = ei[e]; dn = ei[E + e]; } else { s = dn = e - E; }
  float alpha = exv[e * H + hh] / (ssum[dn * H + hh] + EPSF);
  float4 v = *(const float4*)(h + (size_t)s * F + c0);
  float* o = agg + (size_t)dn * F + c0;
  atomicAdd(o + 0, v.x * alpha);
  atomicAdd(o + 1, v.y * alpha);
  atomicAdd(o + 2, v.z * alpha);
  atomicAdd(o + 3, v.w * alpha);
}

// ---------------- bias + ELU fused with bf16 convert (feeds next WMMA GEMM) ----------------
__global__ void k_bias_elu_bf16(const float* __restrict__ agg, const float* __restrict__ bias,
                                unsigned short* __restrict__ out, int total4, int F) {
  int i = blockIdx.x * blockDim.x + threadIdx.x;
  if (i >= total4) return;
  int c = (i * 4) % F;
  float4 v = ((const float4*)agg)[i];
  v.x = eluf(v.x + bias[c + 0]);
  v.y = eluf(v.y + bias[c + 1]);
  v.z = eluf(v.z + bias[c + 2]);
  v.w = eluf(v.w + bias[c + 3]);
  ushort4 o; o.x = f2bf(v.x); o.y = f2bf(v.y); o.z = f2bf(v.z); o.w = f2bf(v.w);
  ((ushort4*)out)[i] = o;
}

// ---------------- head: out[n,:] = elu(agg2[n,:]+b2) @ Wh + bh  (C=32, OUT=4) ----------------
__global__ void k_head(const float* __restrict__ agg2, const float* __restrict__ b2,
                       const float* __restrict__ Wh, const float* __restrict__ bh,
                       float* __restrict__ out, int N) {
  int n = blockIdx.x * blockDim.x + threadIdx.x;
  if (n >= N) return;
  float o0 = bh[0], o1 = bh[1], o2 = bh[2], o3 = bh[3];
  const float* a = agg2 + (size_t)n * 32;
  for (int c = 0; c < 32; ++c) {
    float v = eluf(a[c] + b2[c]);
    const float* w = Wh + c * 4;
    o0 += v * w[0]; o1 += v * w[1]; o2 += v * w[2]; o3 += v * w[3];
  }
  float* op = out + (size_t)n * 4;
  op[0] = o0; op[1] = o1; op[2] = o2; op[3] = o3;
}

extern "C" void kernel_launch(void* const* d_in, const int* in_sizes, int n_in,
                              void* d_out, int out_size, void* d_ws, size_t ws_size,
                              hipStream_t stream) {
  const float* x        = (const float*)d_in[0];
  const int*   ei       = (const int*)d_in[1];
  const float* W1       = (const float*)d_in[2];
  const float* att_src1 = (const float*)d_in[3];
  const float* att_dst1 = (const float*)d_in[4];
  const float* b1       = (const float*)d_in[5];
  const float* W2       = (const float*)d_in[6];
  const float* att_src2 = (const float*)d_in[7];
  const float* att_dst2 = (const float*)d_in[8];
  const float* b2       = (const float*)d_in[9];
  const float* Wh       = (const float*)d_in[10];
  const float* bh       = (const float*)d_in[11];

  const int D = 256, H1 = 8, C1 = 32, C2 = 32;
  const int N = in_sizes[0] / D;          // 50000 (divisible by 16)
  const int E = in_sizes[1] / 2;          // 800000
  const int Etot = E + N;

  // ---- workspace bump allocator (256B aligned) ----
  char* base = (char*)d_ws;
  size_t off = 0;
  auto alloc = [&](size_t bytes) -> void* {
    void* p = base + off;
    off = (off + bytes + 255) & ~(size_t)255;
    return p;
  };
  unsigned short* xbf  = (unsigned short*)alloc((size_t)N * D * 2);
  unsigned short* w1t  = (unsigned short*)alloc((size_t)D * 256 * 2);
  unsigned short* w2t  = (unsigned short*)alloc((size_t)32 * D * 2);
  float*    h1   = (float*)alloc((size_t)N * 256 * 4);
  float*    as1  = (float*)alloc((size_t)N * H1 * 4);
  float*    ad1  = (float*)alloc((size_t)N * H1 * 4);
  float*    s1   = (float*)alloc((size_t)N * H1 * 4);
  unsigned* m1   = (unsigned*)alloc((size_t)N * H1 * 4);
  float*    agg1 = (float*)alloc((size_t)N * 256 * 4);
  float*    ex1  = (float*)alloc((size_t)Etot * H1 * 4);
  float*    h2   = (float*)alloc((size_t)N * C2 * 4);
  float*    as2  = (float*)alloc((size_t)N * 4);
  float*    ad2  = (float*)alloc((size_t)N * 4);
  float*    s2   = (float*)alloc((size_t)N * 4);
  unsigned* m2   = (unsigned*)alloc((size_t)N * 4);
  float*    agg2 = (float*)alloc((size_t)N * C2 * 4);
  float*    ex2  = (float*)alloc((size_t)Etot * 4);
  unsigned short* elubf = xbf;            // reuse: x_bf16 dead after GEMM1

  auto cdiv = [](long long a, long long b) { return (int)((a + b - 1) / b); };
  const int TB = 256;

  // ---- stage 0: conversions ----
  k_f32_to_bf16<<<cdiv((long long)N * D / 4, TB), TB, 0, stream>>>(x, xbf, N * D / 4);
  k_transpose_bf16<<<cdiv(D * 256, TB), TB, 0, stream>>>(W1, w1t, D, 256);
  k_transpose_bf16<<<cdiv(D * 32, TB), TB, 0, stream>>>(W2, w2t, D, 32);

  // ---- layer 1 ----
  {
    int mt = N / 16, nt = 16;
    dim3 blk(32, 8);
    k_wmma_gemm_bf16<<<cdiv((long long)mt * nt, 8), blk, 0, stream>>>(xbf, w1t, h1, mt, nt);
  }
  k_attn_scores<<<cdiv((long long)N * H1, TB), TB, 0, stream>>>(h1, att_src1, att_dst1,
                                                               as1, ad1, N, H1, C1);
  hipMemsetAsync(m1, 0, (size_t)N * H1 * 4, stream);   // encoded 0 < enc(-inf): identity for max
  hipMemsetAsync(s1, 0, (size_t)N * H1 * 4, stream);
  hipMemsetAsync(agg1, 0, (size_t)N * 256 * 4, stream);
  k_edge_max<<<cdiv((long long)Etot * H1, TB), TB, 0, stream>>>(ei, E, N, as1, ad1, m1, H1);
  k_edge_expsum<<<cdiv((long long)Etot * H1, TB), TB, 0, stream>>>(ei, E, N, as1, ad1, m1, s1, ex1, H1);
  k_edge_scatter<<<cdiv((long long)Etot * 64, TB), TB, 0, stream>>>(ei, E, N, ex1, s1, h1, agg1, H1, C1);
  k_bias_elu_bf16<<<cdiv((long long)N * 256 / 4, TB), TB, 0, stream>>>(agg1, b1, elubf, N * 256 / 4, 256);

  // ---- layer 2 ----
  {
    int mt = N / 16, nt = 2;
    dim3 blk(32, 8);
    k_wmma_gemm_bf16<<<cdiv((long long)mt * nt, 8), blk, 0, stream>>>(elubf, w2t, h2, mt, nt);
  }
  k_attn_scores<<<cdiv((long long)N, TB), TB, 0, stream>>>(h2, att_src2, att_dst2,
                                                           as2, ad2, N, 1, C2);
  hipMemsetAsync(m2, 0, (size_t)N * 4, stream);
  hipMemsetAsync(s2, 0, (size_t)N * 4, stream);
  hipMemsetAsync(agg2, 0, (size_t)N * C2 * 4, stream);
  k_edge_max<<<cdiv((long long)Etot, TB), TB, 0, stream>>>(ei, E, N, as2, ad2, m2, 1);
  k_edge_expsum<<<cdiv((long long)Etot, TB), TB, 0, stream>>>(ei, E, N, as2, ad2, m2, s2, ex2, 1);
  k_edge_scatter<<<cdiv((long long)Etot * 8, TB), TB, 0, stream>>>(ei, E, N, ex2, s2, h2, agg2, 1, C2);

  // ---- head ----
  k_head<<<cdiv(N, TB), TB, 0, stream>>>(agg2, b2, Wh, bh, (float*)d_out, N);
}
